// GNNBlock_79559974191618
// MI455X (gfx1250) — matrix operand order repaired and checked by reading, not compiled, not run on hardware
//
#include <hip/hip_runtime.h>
#include <hip/hip_bf16.h>

// ---------------------------------------------------------------------------
// Types
// ---------------------------------------------------------------------------
typedef __attribute__((ext_vector_type(16))) __bf16        v16bf;
typedef __attribute__((ext_vector_type(8)))  float         v8f;
typedef __attribute__((ext_vector_type(4))) unsigned int   uint4v;
typedef __attribute__((ext_vector_type(2))) unsigned int   uint2v;
typedef __attribute__((ext_vector_type(4)))  float         float4v;
typedef __attribute__((ext_vector_type(4)))  int           int4v;

union ABFrag { struct { uint4v lo; uint4v hi; } u; v16bf v; };

#define NEG_BIG (-3.402823466e+38f)

#if __has_builtin(__builtin_amdgcn_global_load_async_to_lds_b128)
#define GAT_ASYNC 1
#define AS1 __attribute__((address_space(1)))
#define AS3 __attribute__((address_space(3)))
#define GPTR(p) ((AS1 int4v*)(p))
#define LPTR(p) ((AS3 int4v*)(p))
#else
#define GAT_ASYNC 0
#endif

static __device__ __forceinline__ void wait_async0() {
#if __has_builtin(__builtin_amdgcn_s_wait_asynccnt)
    __builtin_amdgcn_s_wait_asynccnt(0);
#else
    asm volatile("s_wait_asynccnt 0x0" ::: "memory");
#endif
}

// fp32 -> bf16 (round-to-nearest-even)
static __device__ __forceinline__ unsigned short f2bf(float f) {
    unsigned u = __float_as_uint(f);
    u += 0x7FFFu + ((u >> 16) & 1u);
    return (unsigned short)(u >> 16);
}
static __device__ __forceinline__ unsigned pk2bf(float a, float b) {
    return (unsigned)f2bf(a) | ((unsigned)f2bf(b) << 16);
}

static __device__ __forceinline__ void get_edge(const int* __restrict__ ei,
                                                int E, int e, int& s, int& d) {
    if (e < E) { s = ei[e]; d = ei[E + e]; }
    else       { s = e - E; d = e - E; }          // self loops appended
}

// ---------------------------------------------------------------------------
// Bulk fp32 -> bf16 conversion (row-major, flat). 8 elems / thread.
// ---------------------------------------------------------------------------
__global__ __launch_bounds__(256)
void k_cvt_rows(const float* __restrict__ src, unsigned short* __restrict__ dst,
                long long count)
{
    long long i = ((long long)blockIdx.x * 256 + threadIdx.x) * 8;
    if (i >= count) return;
    float4v a = *(const float4v*)(src + i);
    float4v b = *(const float4v*)(src + i + 4);
    uint4v pk;
    pk.x = pk2bf(a.x, a.y); pk.y = pk2bf(a.z, a.w);
    pk.z = pk2bf(b.x, b.y); pk.w = pk2bf(b.z, b.w);
    *(uint4v*)(dst + i) = pk;
}

// ---------------------------------------------------------------------------
// W[K x Nn] fp32 -> WT[Nn x K] bf16 (transpose + convert), 32x32 LDS tiles.
// grid = (Nn/32, K/32), block = 256
// ---------------------------------------------------------------------------
#define TS 36   // padded LDS tile stride (bf16 elems)

__global__ __launch_bounds__(256)
void k_cvt_wT(const float* __restrict__ W, unsigned short* __restrict__ WT,
              int K, int Nn)
{
    __shared__ unsigned short tile[32 * TS];
    const int t  = threadIdx.x;
    const int nb = blockIdx.x * 32;
    const int kb = blockIdx.y * 32;

    {   // load 32(k) x 32(n) fp32, convert, store to LDS [k][n]
        int kr  = t >> 3;           // 0..31
        int nc  = (t & 7) << 2;     // 0..28 step 4
        float4v v = *(const float4v*)(W + (size_t)(kb + kr) * Nn + nb + nc);
        uint2v pk; pk.x = pk2bf(v.x, v.y); pk.y = pk2bf(v.z, v.w);
        *(uint2v*)&tile[kr * TS + nc] = pk;     // 8B aligned (72*kr + 8*j)
    }
    __syncthreads();
    {   // write transposed: WT[n][k], 4 k-values per thread
        int nr = t >> 3;            // 0..31
        int kc = (t & 7) << 2;      // 0..28 step 4
        unsigned short e0 = tile[(kc + 0) * TS + nr];
        unsigned short e1 = tile[(kc + 1) * TS + nr];
        unsigned short e2 = tile[(kc + 2) * TS + nr];
        unsigned short e3 = tile[(kc + 3) * TS + nr];
        uint2v pk;
        pk.x = (unsigned)e0 | ((unsigned)e1 << 16);
        pk.y = (unsigned)e2 | ((unsigned)e3 << 16);
        *(uint2v*)(WT + (size_t)(nb + nr) * K + kb + kc) = pk;
    }
}

// ---------------------------------------------------------------------------
// Tiled bf16 WMMA GEMM:  H[M x Nn] = Xb[M x K] * WT[Nn x K]^T  (bf16 in, f32 out)
// Both operands row-major bf16 with K contiguous -> staging is a pure copy.
// Block = 256 thr (8 waves). Tile 128x128, K-step 32. Wave tile 32x64.
// Double-buffered LDS; staging via async global->LDS when available.
// ---------------------------------------------------------------------------
#define BM 128
#define BN 128
#define BK 32
#define LDA 40   // padded LDS row stride (bf16) for A (M-major, K contig)
#define LDB 40   // padded LDS row stride for B (N-major, K contig)

__global__ __launch_bounds__(256)
void k_gemm_bf16(const unsigned short* __restrict__ Xb,
                 const unsigned short* __restrict__ WT,
                 float* __restrict__ H, int K, int Nn)
{
    __shared__ unsigned short sA[2][BM * LDA];
    __shared__ unsigned short sB[2][BN * LDB];

    const int tid   = threadIdx.x;
    const int lane  = tid & 31;
    const int wave  = tid >> 5;
    const int hlf   = lane >> 4;      // 0: lanes 0-15, 1: lanes 16-31
    const int l16   = lane & 15;
    const int bm    = blockIdx.y * BM;
    const int bn    = blockIdx.x * BN;
    const int waveM = (wave & 3) * 32;   // 4 waves along M
    const int waveN = (wave >> 2) * 64;  // 2 waves along N

    // staging: thread t copies one 16B chunk of A and one of B per K-step
    const int srow = tid >> 1;           // 0..127
    const int shb  = (tid & 1) << 4;     // 0 or 16 (bf16 elems)
    const unsigned short* gA = Xb + (size_t)(bm + srow) * K + shb;
    const unsigned short* gB = WT + (size_t)(bn + srow) * K + shb;

    v8f acc[2][4];
#pragma unroll
    for (int mi = 0; mi < 2; ++mi)
#pragma unroll
        for (int ni = 0; ni < 4; ++ni)
#pragma unroll
            for (int r = 0; r < 8; ++r) acc[mi][ni][r] = 0.0f;

#if !GAT_ASYNC
    uint4v areg, breg;
#endif

    auto issue_tile = [&](int k0, int buf) {
#if GAT_ASYNC
        __builtin_amdgcn_global_load_async_to_lds_b128(
            GPTR(gA + k0), LPTR(&sA[buf][srow * LDA + shb]), 0, 0);
        __builtin_amdgcn_global_load_async_to_lds_b128(
            GPTR(gB + k0), LPTR(&sB[buf][srow * LDB + shb]), 0, 0);
#else
        (void)buf;
        areg = *(const uint4v*)(gA + k0);
        breg = *(const uint4v*)(gB + k0);
#endif
    };
    auto commit_tile = [&](int buf) {
#if GAT_ASYNC
        (void)buf;
        wait_async0();                 // LDS writes of our async loads done
#else
        *(uint4v*)&sA[buf][srow * LDA + shb] = areg;   // 80*row + 32*h : 16B aligned
        *(uint4v*)&sB[buf][srow * LDB + shb] = breg;
#endif
    };
    auto compute = [&](int cur) {
        // A 16x32: lanes 0-15 row M=l16 hold K 0-7 & 16-23; lanes 16-31: K 8-15 & 24-31
        ABFrag afr[2];
#pragma unroll
        for (int mi = 0; mi < 2; ++mi) {
            int row = waveM + mi * 16 + l16;
            int klo = hlf * 8;
            afr[mi].u.lo = *(const uint4v*)&sA[cur][row * LDA + klo];
            afr[mi].u.hi = *(const uint4v*)&sA[cur][row * LDA + klo + 16];
        }
        // B 32x16: lanes 0-15 col N=l16 hold K 0-7 & 16-23; lanes 16-31: K 8-15 & 24-31
        ABFrag bfr[4];
#pragma unroll
        for (int ni = 0; ni < 4; ++ni) {
            int n   = waveN + ni * 16 + l16;
            int kst = hlf * 8;
            bfr[ni].u.lo = *(const uint4v*)&sB[cur][n * LDB + kst];
            bfr[ni].u.hi = *(const uint4v*)&sB[cur][n * LDB + kst + 16];
        }
#pragma unroll
        for (int mi = 0; mi < 2; ++mi)
#pragma unroll
            for (int ni = 0; ni < 4; ++ni)
                acc[mi][ni] = __builtin_amdgcn_wmma_f32_16x16x32_bf16(
                    false, afr[mi].v, false, bfr[ni].v,
                    (short)0, acc[mi][ni], false, false);
    };

    const int nK = K / BK;

    issue_tile(0, 0);
    commit_tile(0);
    __syncthreads();

    for (int kt = 0; kt + 1 < nK; ++kt) {          // steady state: branch-free staging
        const int cur = kt & 1;
        issue_tile((kt + 1) * BK, cur ^ 1);        // next buffer free since last barrier
        const int pkt = (kt + 2 < nK) ? kt + 2 : nK - 1;   // clamped (select, no branch)
        __builtin_prefetch(gA + pkt * BK, 0, 3);
        __builtin_prefetch(gB + pkt * BK, 0, 3);
        compute(cur);
        commit_tile(cur ^ 1);
        __syncthreads();
    }
    compute((nK - 1) & 1);                          // last tile, nothing left to stage

    // ---- epilogue: C/D layout -> fp32 H (lane<16: M=r, lane>=16: M=r+8) ----
#pragma unroll
    for (int mi = 0; mi < 2; ++mi) {
        int row0 = bm + waveM + mi * 16 + hlf * 8;
#pragma unroll
        for (int ni = 0; ni < 4; ++ni) {
            int col = bn + waveN + ni * 16 + l16;
            float* hp = H + (size_t)row0 * Nn + col;
#pragma unroll
            for (int r = 0; r < 8; ++r) hp[(size_t)r * Nn] = acc[mi][ni][r];
        }
    }
}

// ---------------------------------------------------------------------------
// Per-node attention scores: s_src[n] = h[n].a_src, s_dst[n] = h[n].a_dst
// ---------------------------------------------------------------------------
__global__ __launch_bounds__(256)
void k_scores(const float* __restrict__ H, const float* __restrict__ asr,
              const float* __restrict__ ads, float* __restrict__ s_src,
              float* __restrict__ s_dst, int Fo)
{
    __shared__ float red0[256];
    __shared__ float red1[256];
    const int n = blockIdx.x;
    const float* h = H + (size_t)n * Fo;
    float ss = 0.f, sd = 0.f;
    for (int f = threadIdx.x; f < Fo; f += 256) {
        float hv = h[f];
        ss += hv * asr[f];
        sd += hv * ads[f];
    }
    red0[threadIdx.x] = ss; red1[threadIdx.x] = sd;
    __syncthreads();
    for (int s = 128; s > 0; s >>= 1) {
        if (threadIdx.x < (unsigned)s) {
            red0[threadIdx.x] += red0[threadIdx.x + s];
            red1[threadIdx.x] += red1[threadIdx.x + s];
        }
        __syncthreads();
    }
    if (threadIdx.x == 0) { s_src[n] = red0[0]; s_dst[n] = red1[0]; }
}

// ---------------------------------------------------------------------------
// CSR-by-dst build (once per launch; row-sorted => deterministic)
// ---------------------------------------------------------------------------
__global__ void k_zero_u32(unsigned* p, int n) {
    int i = blockIdx.x * blockDim.x + threadIdx.x;
    if (i < n) p[i] = 0u;
}

__global__ void k_count(const int* __restrict__ ei, int E, int Etot,
                        unsigned* __restrict__ counts) {
    int e = blockIdx.x * blockDim.x + threadIdx.x;
    if (e >= Etot) return;
    int s, d; get_edge(ei, E, e, s, d);
    atomicAdd(&counts[d], 1u);
}

__global__ void k_scan(const unsigned* __restrict__ counts,
                       unsigned* __restrict__ row_ptr,
                       unsigned* __restrict__ cursor, int Nn) {
    __shared__ unsigned part[256];
    const int chunk = Nn / 256;     // 16
    unsigned s = 0;
    for (int i = 0; i < chunk; ++i) s += counts[threadIdx.x * chunk + i];
    part[threadIdx.x] = s;
    __syncthreads();
    if (threadIdx.x == 0) {
        unsigned acc = 0;
        for (int i = 0; i < 256; ++i) { unsigned t = part[i]; part[i] = acc; acc += t; }
        row_ptr[Nn] = acc;
    }
    __syncthreads();
    unsigned acc = part[threadIdx.x];
    for (int i = 0; i < chunk; ++i) {
        int idx = threadIdx.x * chunk + i;
        row_ptr[idx] = acc; cursor[idx] = acc;
        acc += counts[idx];
    }
}

__global__ void k_scatter(const int* __restrict__ ei, int E, int Etot,
                          unsigned* __restrict__ cursor,
                          int* __restrict__ csr_src, int* __restrict__ csr_eid) {
    int e = blockIdx.x * blockDim.x + threadIdx.x;
    if (e >= Etot) return;
    int s, d; get_edge(ei, E, e, s, d);
    unsigned p = atomicAdd(&cursor[d], 1u);
    csr_src[p] = s; csr_eid[p] = e;
}

__global__ void k_sortrows(const unsigned* __restrict__ row_ptr,
                           int* __restrict__ csr_src, int* __restrict__ csr_eid, int Nn) {
    int d = blockIdx.x * blockDim.x + threadIdx.x;
    if (d >= Nn) return;
    int beg = (int)row_ptr[d], end = (int)row_ptr[d + 1];
    for (int i = beg + 1; i < end; ++i) {            // insertion sort by edge id
        int ke = csr_eid[i], ks = csr_src[i];
        int j = i - 1;
        while (j >= beg && csr_eid[j] > ke) {
            csr_eid[j + 1] = csr_eid[j]; csr_src[j + 1] = csr_src[j]; --j;
        }
        csr_eid[j + 1] = ke; csr_src[j + 1] = ks;
    }
}

// ---------------------------------------------------------------------------
// Edge scores + per-dst softmax (no atomics -> deterministic)
// ---------------------------------------------------------------------------
__global__ void k_edge_score(const int* __restrict__ ei, int E, int Etot,
                             const float* __restrict__ s_src,
                             const float* __restrict__ s_dst,
                             float* __restrict__ escore) {
    int e = blockIdx.x * blockDim.x + threadIdx.x;
    if (e >= Etot) return;
    int s, d; get_edge(ei, E, e, s, d);
    float v = s_src[s] + s_dst[d];
    escore[e] = (v > 0.f) ? v : 0.2f * v;           // leaky_relu(0.2)
}

__global__ void k_softmax(const float* __restrict__ escore,
                          const unsigned* __restrict__ row_ptr,
                          const int* __restrict__ csr_eid,
                          float* __restrict__ alpha, int Nn) {
    int d = blockIdx.x * blockDim.x + threadIdx.x;
    if (d >= Nn) return;
    unsigned beg = row_ptr[d], end = row_ptr[d + 1];
    float m = NEG_BIG;
    for (unsigned p = beg; p < end; ++p) m = fmaxf(m, escore[csr_eid[p]]);
    float den = 0.f;
    for (unsigned p = beg; p < end; ++p) {
        float w = __expf(escore[csr_eid[p]] - m);
        alpha[p] = w; den += w;
    }
    float inv = 1.f / (den + 1e-16f);
    for (unsigned p = beg; p < end; ++p) alpha[p] *= inv;
}

// ---------------------------------------------------------------------------
// y[d,f] = sum_p alpha[p]*h[src[p],f] + b[f]
// layers 1-3: y = leaky_relu(y, .01) stored as bf16 (next GEMM input)
// layer 4:    y stored fp32 to d_out
// grid = (N, Fo/256)
// ---------------------------------------------------------------------------
__global__ __launch_bounds__(256)
void k_aggregate(const float* __restrict__ H, const float* __restrict__ alpha,
                 const unsigned* __restrict__ row_ptr, const int* __restrict__ csr_src,
                 const float* __restrict__ bias, float* __restrict__ outf,
                 unsigned short* __restrict__ outb, int Fo, int act)
{
    const int d = blockIdx.x;
    const int f = blockIdx.y * 256 + threadIdx.x;
    const unsigned beg = row_ptr[d], end = row_ptr[d + 1];
    float acc = 0.f;
    for (unsigned p = beg; p < end; ++p) {
        int s = csr_src[p];
        acc += alpha[p] * H[(size_t)s * Fo + f];
    }
    acc += bias[f];
    if (act) {
        acc = (acc > 0.f) ? acc : 0.01f * acc;
        outb[(size_t)d * Fo + f] = f2bf(acc);
    } else {
        outf[(size_t)d * Fo + f] = acc;
    }
}

// ---------------------------------------------------------------------------
// Driver
// ---------------------------------------------------------------------------
extern "C" void kernel_launch(void* const* d_in, const int* in_sizes, int n_in,
                              void* d_out, int out_size, void* d_ws, size_t ws_size,
                              hipStream_t stream)
{
    (void)in_sizes; (void)n_in; (void)out_size; (void)ws_size;
    const int N = 4096, E = 32768, Etot = E + N;

    const float* x  = (const float*)d_in[0];
    const int*   ei = (const int*)d_in[1];

    char* wsp = (char*)d_ws; size_t off = 0;
    auto carve = [&](size_t bytes) -> char* {
        char* p = wsp + off; off += (bytes + 255) & ~(size_t)255; return p;
    };
    float*          bufH    = (float*)carve((size_t)N * 8192 * 4);          // h (max layer)
    unsigned short* bufXb   = (unsigned short*)carve((size_t)N * 4096 * 2); // bf16 activations
    unsigned short* bufWT   = (unsigned short*)carve((size_t)8192 * 4096 * 2); // bf16 W^T
    float*    s_src   = (float*)carve((size_t)N * 4);
    float*    s_dst   = (float*)carve((size_t)N * 4);
    float*    escore  = (float*)carve((size_t)Etot * 4);
    float*    alphab  = (float*)carve((size_t)Etot * 4);
    unsigned* counts  = (unsigned*)carve((size_t)N * 4);
    unsigned* row_ptr = (unsigned*)carve((size_t)(N + 1) * 4);
    unsigned* cursor  = (unsigned*)carve((size_t)N * 4);
    int*      csr_src = (int*)carve((size_t)Etot * 4);
    int*      csr_eid = (int*)carve((size_t)Etot * 4);

    // ---- CSR by destination (shared across layers) ----
    k_zero_u32<<<(N + 255) / 256, 256, 0, stream>>>(counts, N);
    k_count   <<<(Etot + 255) / 256, 256, 0, stream>>>(ei, E, Etot, counts);
    k_scan    <<<1, 256, 0, stream>>>(counts, row_ptr, cursor, N);
    k_scatter <<<(Etot + 255) / 256, 256, 0, stream>>>(ei, E, Etot, cursor, csr_src, csr_eid);
    k_sortrows<<<(N + 255) / 256, 256, 0, stream>>>(row_ptr, csr_src, csr_eid, N);

    const int FI[4] = { 768, 1024, 2048, 4096 };
    const int FO[4] = { 1024, 2048, 4096, 8192 };

    // first layer input -> bf16
    {
        long long cnt = (long long)N * FI[0];
        k_cvt_rows<<<(unsigned)((cnt / 8 + 255) / 256), 256, 0, stream>>>(x, bufXb, cnt);
    }

    for (int l = 0; l < 4; ++l) {
        const float* W   = (const float*)d_in[2 + 4 * l];
        const float* asr = (const float*)d_in[3 + 4 * l];
        const float* ads = (const float*)d_in[4 + 4 * l];
        const float* bia = (const float*)d_in[5 + 4 * l];

        // W -> bf16 transposed [Fo x Fi]
        dim3 gt(FO[l] / 32, FI[l] / 32);
        k_cvt_wT<<<gt, 256, 0, stream>>>(W, bufWT, FI[l], FO[l]);

        dim3 gg(FO[l] / BN, N / BM);
        k_gemm_bf16<<<gg, 256, 0, stream>>>(bufXb, bufWT, bufH, FI[l], FO[l]);

        k_scores    <<<N, 256, 0, stream>>>(bufH, asr, ads, s_src, s_dst, FO[l]);
        k_edge_score<<<(Etot + 255) / 256, 256, 0, stream>>>(ei, E, Etot, s_src, s_dst, escore);
        k_softmax   <<<(N + 255) / 256, 256, 0, stream>>>(escore, row_ptr, csr_eid, alphab, N);

        dim3 ga(N, FO[l] / 256);
        k_aggregate<<<ga, 256, 0, stream>>>(bufH, alphab, row_ptr, csr_src, bia,
                                            (float*)d_out, bufXb, FO[l], (l < 3) ? 1 : 0);
    }
}
